// Attention1d_76089640616293
// MI455X (gfx1250) — matrix operand order repaired and checked by reading, not compiled, not run on hardware
//
#include <hip/hip_runtime.h>
#include <hip/hip_bf16.h>

// ---------------------------------------------------------------------------
// Problem constants (reference: B=8, C=512, T=2048, NHEAD=8, GROUPS=32)
// ---------------------------------------------------------------------------
#define BATCH 8
#define CCH   512
#define TT    2048
#define NHEAD_ 8
#define GROUPS_ 32
#define CPG   16          // channels per group
#define W3C   1536        // 3*C
#define HD    64          // head dim

typedef __bf16 bf16;
typedef __attribute__((ext_vector_type(8)))  bf16  v8bf;
typedef __attribute__((ext_vector_type(16))) bf16  v16bf;
typedef __attribute__((ext_vector_type(8)))  float v8f;

// ---------------------------------------------------------------------------
// WMMA helpers (CDNA5 bf16 16x16x32, fp32 accumulate)
// A-matrix (16x32, 16-bit): lane l row m=l%16; elems 0..7 -> K=(l/16)*8+0..7,
//                           elems 8..15 -> K=(l/16)*8+16..23
// B-matrix (32x16, 16-bit): lane l col n=l%16; elems 0..7 -> K=(l/16)*8+0..7,
//                           elems 8..15 -> K=(l/16)*8+16..23
// C/D (16x16 f32):          lane l col n=l%16; vgpr j -> M=(l/16)*8+j
// ---------------------------------------------------------------------------
__device__ __forceinline__ v16bf concat8(v8bf lo, v8bf hi) {
  v16bf r;
#pragma unroll
  for (int i = 0; i < 8; ++i) { r[i] = lo[i]; r[i + 8] = hi[i]; }
  return r;
}

// rowk = &A[(m0 + lane%16)*lda + k0] ; row-major A, 32 consecutive K
__device__ __forceinline__ v16bf load_afrag(const bf16* rowk) {
  const int half = (threadIdx.x & 31) >> 4;
  v8bf lo = *(const v8bf*)(rowk + half * 8);
  v8bf hi = *(const v8bf*)(rowk + half * 8 + 16);
  return concat8(lo, hi);
}

// rowk = &Brow_n[k0] ; row-major in K for column n = lane%16
__device__ __forceinline__ v16bf load_bfrag(const bf16* rowk) {
  const int half = (threadIdx.x & 31) >> 4;
  v8bf lo = *(const v8bf*)(rowk + half * 8);
  v8bf hi = *(const v8bf*)(rowk + half * 8 + 16);
  return concat8(lo, hi);
}

__device__ __forceinline__ v8f wmma_bf16(v16bf a, v16bf b, v8f c) {
  return __builtin_amdgcn_wmma_f32_16x16x32_bf16(
      /*neg_a=*/false, a, /*neg_b=*/false, b,
      /*c_mod=*/(short)0, c, /*reuse_a=*/false, /*reuse_b=*/false);
}

// ---------------------------------------------------------------------------
// Kernel 1: fp32 -> bf16 elementwise (weights)
// ---------------------------------------------------------------------------
__global__ __launch_bounds__(256) void f32_to_bf16_kernel(const float* __restrict__ in,
                                                          bf16* __restrict__ out, int n) {
  int i = blockIdx.x * 256 + threadIdx.x;
  if (i < n) out[i] = (bf16)in[i];
}

// ---------------------------------------------------------------------------
// Kernel 2: GroupNorm, fp32 in (either (B,C,T) or (B,T,C)) -> bf16 (B,T,C)
// one block per (b, group); 16 channels x 2048 cols reduced in fp32
// ---------------------------------------------------------------------------
__global__ __launch_bounds__(256) void groupnorm_kernel(const float* __restrict__ in,
                                                        const float* __restrict__ gamma,
                                                        const float* __restrict__ beta,
                                                        bf16* __restrict__ outTC,
                                                        int in_is_ct) {
  __shared__ float s_a[256];
  __shared__ float s_b[256];
  const int tid = threadIdx.x;
  const int b = blockIdx.x >> 5;       // / GROUPS
  const int g = blockIdx.x & 31;       // % GROUPS
  const int N = CPG * TT;              // 32768 elements per group

  float sum = 0.f, sq = 0.f;
  for (int e = tid; e < N; e += 256) {
    int ci, t; size_t idx;
    if (in_is_ct) { ci = e >> 11; t = e & (TT - 1);
                    idx = ((size_t)b * CCH + g * CPG + ci) * TT + t; }
    else          { t = e >> 4; ci = e & (CPG - 1);
                    idx = ((size_t)b * TT + t) * CCH + g * CPG + ci; }
    float v = in[idx];
    sum += v; sq += v * v;
  }
  s_a[tid] = sum; s_b[tid] = sq;
  __syncthreads();
  for (int s = 128; s > 0; s >>= 1) {
    if (tid < s) { s_a[tid] += s_a[tid + s]; s_b[tid] += s_b[tid + s]; }
    __syncthreads();
  }
  const float mean = s_a[0] * (1.0f / N);
  const float var  = s_b[0] * (1.0f / N) - mean * mean;
  const float rstd = rsqrtf(var + 1e-5f);

  for (int e = tid; e < N; e += 256) {
    int ci, t; size_t idx;
    if (in_is_ct) { ci = e >> 11; t = e & (TT - 1);
                    idx = ((size_t)b * CCH + g * CPG + ci) * TT + t; }
    else          { t = e >> 4; ci = e & (CPG - 1);
                    idx = ((size_t)b * TT + t) * CCH + g * CPG + ci; }
    const int c = g * CPG + ci;
    float v = (in[idx] - mean) * rstd * gamma[c] + beta[c];
    outTC[((size_t)b * TT + t) * CCH + c] = (bf16)v;
  }
}

// ---------------------------------------------------------------------------
// Kernel 3: WMMA GEMM.  D[b][m][n] = sum_k W[m][k] * X[b][n][k] + bias[m]
//   W: (M,K) bf16 row-major.  X: (B,N,K) bf16.
//   outTC != nullptr : store bf16 at outTC[(b*N+n)*M + m]          (QKV path)
//   outCT != nullptr : store f32  at outCT[(b*M+m)*N + n] + resid  (proj path)
// wave tile 16M x 64N; block tile 128M x 64N (8 waves); K-step 32.
// Software pipelined: next k-step's fragments prefetched into a second
// register set while the current 4 WMMAs issue (hides one load latency).
// ---------------------------------------------------------------------------
__global__ __launch_bounds__(256) void gemm_bf16_kernel(const bf16* __restrict__ W,
                                                        const bf16* __restrict__ X,
                                                        const float* __restrict__ bias,
                                                        bf16* __restrict__ outTC,
                                                        float* __restrict__ outCT,
                                                        const float* __restrict__ residCT,
                                                        int M, int N, int K) {
  const int lane = threadIdx.x & 31;
  const int wv   = threadIdx.x >> 5;
  const int bb   = blockIdx.z;
  const int m0   = blockIdx.x * 128 + wv * 16;
  const int n0   = blockIdx.y * 64;
  if (m0 >= M) return;
  const int half = lane >> 4;
  const int nlo  = lane & 15;

  v8f acc[4];
#pragma unroll
  for (int ct = 0; ct < 4; ++ct)
#pragma unroll
    for (int j = 0; j < 8; ++j) acc[ct][j] = 0.f;

  const bf16* arow  = W + (size_t)(m0 + nlo) * K;
  const bf16* xbase = X + ((size_t)bb * N + n0) * K;

  // prologue: k-step 0 fragments
  v16bf a_cur = load_afrag(arow);
  v16bf b_cur[4];
#pragma unroll
  for (int nt = 0; nt < 4; ++nt)
    b_cur[nt] = load_bfrag(xbase + (size_t)(nt * 16 + nlo) * K);

#pragma unroll 2
  for (int k0 = 32; k0 < K; k0 += 32) {
    // prefetch next k-step into distinct registers
    v16bf a_nxt = load_afrag(arow + k0);
    v16bf b_nxt[4];
#pragma unroll
    for (int nt = 0; nt < 4; ++nt)
      b_nxt[nt] = load_bfrag(xbase + (size_t)(nt * 16 + nlo) * K + k0);
    // compute on current fragments while prefetch is in flight
#pragma unroll
    for (int nt = 0; nt < 4; ++nt)
      acc[nt] = wmma_bf16(a_cur, b_cur[nt], acc[nt]);
    a_cur = a_nxt;
#pragma unroll
    for (int nt = 0; nt < 4; ++nt) b_cur[nt] = b_nxt[nt];
  }
  // epilogue k-step
#pragma unroll
  for (int nt = 0; nt < 4; ++nt)
    acc[nt] = wmma_bf16(a_cur, b_cur[nt], acc[nt]);

#pragma unroll
  for (int nt = 0; nt < 4; ++nt)
#pragma unroll
    for (int j = 0; j < 8; ++j) {
      const int m = m0 + half * 8 + j;
      const int n = n0 + nt * 16 + nlo;
      const float v = acc[nt][j] + bias[m];
      if (outTC) outTC[((size_t)bb * N + n) * M + m] = (bf16)v;
      if (outCT) {
        const size_t o = ((size_t)bb * M + m) * N + n;
        outCT[o] = v + residCT[o];
      }
    }
}

// ---------------------------------------------------------------------------
// Kernel 4: flash attention.  qkv: (B,T,1536) bf16.  outT: (B,T,512) f32.
// block = one (b,h) x 128 query rows; 8 waves, each owns 16 rows, sweeps all
// 2048 keys in steps of 32 with online softmax (scale^2 = 1/8).
// ---------------------------------------------------------------------------
__global__ __launch_bounds__(256) void attn_kernel(const bf16* __restrict__ qkv,
                                                   float* __restrict__ outT) {
  __shared__ __align__(16) bf16 v_lds[HD * 32];      // [c][s] transposed V tile
  __shared__ __align__(16) bf16 p_lds[8][16 * 32];   // per-wave P tile (16 x 32)

  const int lane = threadIdx.x & 31;
  const int wv   = threadIdx.x >> 5;
  const int b    = blockIdx.x >> 3;
  const int h    = blockIdx.x & 7;
  const int t0   = blockIdx.y * 128 + wv * 16;
  const int offq = h * 192, offk = offq + 64, offv = offq + 128;
  const int half = lane >> 4;
  const int nlo  = lane & 15;

  // Q fragments for this wave's 16 rows (K = head channels 0..63)
  const bf16* qrow = qkv + ((size_t)b * TT + t0 + nlo) * W3C + offq;
  const v16bf aq0 = load_afrag(qrow);        // c 0..31
  const v16bf aq1 = load_afrag(qrow + 32);   // c 32..63

  float mrun[8], lrun[8];
  v8f acc[4];
#pragma unroll
  for (int j = 0; j < 8; ++j) { mrun[j] = -1e30f; lrun[j] = 0.f; }
#pragma unroll
  for (int ct = 0; ct < 4; ++ct)
#pragma unroll
    for (int j = 0; j < 8; ++j) acc[ct][j] = 0.f;

  for (int s0 = 0; s0 < TT; s0 += 32) {
    // ---- issue all K-fragment loads for this s-step up front ----
    v16bf bk[2][2];
#pragma unroll
    for (int sh = 0; sh < 2; ++sh) {
      const bf16* krow = qkv + ((size_t)b * TT + s0 + sh * 16 + nlo) * W3C + offk;
      bk[sh][0] = load_bfrag(krow);
      bk[sh][1] = load_bfrag(krow + 32);
    }

    __syncthreads();   // previous iteration's v_lds reads complete
    {  // cooperative V tile load, transposed to [c][s]
      const int sl = threadIdx.x >> 3;        // 0..31 key within tile
      const int c0 = (threadIdx.x & 7) * 8;   // 8 channels per thread
      const v8bf vv = *(const v8bf*)(qkv + ((size_t)b * TT + s0 + sl) * W3C + offv + c0);
#pragma unroll
      for (int i = 0; i < 8; ++i) v_lds[(c0 + i) * 32 + sl] = vv[i];
    }

    // ---- S = (q*scale) . (k*scale) for 32 keys (two 16x16 tiles) ----
    v8f st[2];
#pragma unroll
    for (int sh = 0; sh < 2; ++sh) {
      v8f s_ = {};
      s_ = wmma_bf16(aq0, bk[sh][0], s_);
      s_ = wmma_bf16(aq1, bk[sh][1], s_);
#pragma unroll
      for (int j = 0; j < 8; ++j) st[sh][j] = s_[j] * 0.125f;
    }

    // ---- online softmax (row reductions across the 16-lane half) ----
    float mt[8];
#pragma unroll
    for (int j = 0; j < 8; ++j) {
      float v = fmaxf(st[0][j], st[1][j]);
      for (int off = 1; off < 16; off <<= 1) v = fmaxf(v, __shfl_xor(v, off, 32));
      mt[j] = v;
    }
    float p0[8], p1[8];
#pragma unroll
    for (int j = 0; j < 8; ++j) {
      const float mn    = fmaxf(mrun[j], mt[j]);
      const float alpha = __expf(mrun[j] - mn);
      p0[j] = __expf(st[0][j] - mn);
      p1[j] = __expf(st[1][j] - mn);
      float r = p0[j] + p1[j];
      for (int off = 1; off < 16; off <<= 1) r += __shfl_xor(r, off, 32);
      lrun[j] = lrun[j] * alpha + r;
      mrun[j] = mn;
#pragma unroll
      for (int ct = 0; ct < 4; ++ct) acc[ct][j] *= alpha;
    }

    // ---- P through LDS (C-layout -> A-fragment layout) ----
#pragma unroll
    for (int j = 0; j < 8; ++j) {
      p_lds[wv][(half * 8 + j) * 32 + nlo]      = (bf16)p0[j];
      p_lds[wv][(half * 8 + j) * 32 + 16 + nlo] = (bf16)p1[j];
    }
    __syncthreads();   // V tile in LDS complete (also orders p_lds RAW)
    asm volatile("s_wait_dscnt 0" ::: "memory");   // per-wave RAW on p_lds

    // ---- O += P . V^T : gather all fragments, then 4 WMMAs ----
    const v16bf ap = load_afrag(&p_lds[wv][nlo * 32]);
    v16bf bv[4];
#pragma unroll
    for (int ct = 0; ct < 4; ++ct)
      bv[ct] = load_bfrag(&v_lds[(ct * 16 + nlo) * 32]);
#pragma unroll
    for (int ct = 0; ct < 4; ++ct)
      acc[ct] = wmma_bf16(ap, bv[ct], acc[ct]);
  }

  // ---- normalize + store (B,T,C) fp32 ----
#pragma unroll
  for (int ct = 0; ct < 4; ++ct)
#pragma unroll
    for (int j = 0; j < 8; ++j) {
      const int t = t0 + half * 8 + j;
      const int c = h * HD + ct * 16 + nlo;
      outT[((size_t)b * TT + t) * CCH + c] = acc[ct][j] / lrun[j];
    }
}

// ---------------------------------------------------------------------------
// Host-side orchestration
// ---------------------------------------------------------------------------
extern "C" void kernel_launch(void* const* d_in, const int* in_sizes, int n_in,
                              void* d_out, int out_size, void* d_ws, size_t ws_size,
                              hipStream_t stream) {
  const float* x     = (const float*)d_in[0];
  const float* g1    = (const float*)d_in[1];
  const float* be1   = (const float*)d_in[2];
  const float* wqkv  = (const float*)d_in[3];
  const float* bqkv  = (const float*)d_in[4];
  const float* g2    = (const float*)d_in[5];
  const float* be2   = (const float*)d_in[6];
  const float* wproj = (const float*)d_in[7];
  const float* bproj = (const float*)d_in[8];
  float* out = (float*)d_out;

  char* ws = (char*)d_ws;
  bf16*  xn    = (bf16*)(ws);                                // 16 MB (B,T,C) bf16 (reused)
  bf16*  qkvT  = (bf16*)(ws + (size_t)16 * 1024 * 1024);     // 48 MB (B,T,3C) bf16
  float* aT    = (float*)(ws + (size_t)64 * 1024 * 1024);    // 32 MB (B,T,C) f32
  bf16*  wq16  = (bf16*)(ws + (size_t)96 * 1024 * 1024);     // 1.5 MB
  bf16*  wp16  = (bf16*)(ws + (size_t)98 * 1024 * 1024);     // 0.5 MB

  // weights -> bf16
  f32_to_bf16_kernel<<<(W3C * CCH + 255) / 256, 256, 0, stream>>>(wqkv, wq16, W3C * CCH);
  f32_to_bf16_kernel<<<(CCH * CCH + 255) / 256, 256, 0, stream>>>(wproj, wp16, CCH * CCH);

  // GN1: (B,C,T) f32 -> (B,T,C) bf16
  groupnorm_kernel<<<BATCH * GROUPS_, 256, 0, stream>>>(x, g1, be1, xn, /*in_is_ct=*/1);

  // QKV GEMM: (1536x512) x (B,2048,512) -> (B,2048,1536) bf16
  gemm_bf16_kernel<<<dim3(W3C / 128, TT / 64, BATCH), 256, 0, stream>>>(
      wq16, xn, bqkv, qkvT, nullptr, nullptr, W3C, TT, CCH);

  // attention: (B,T,1536) bf16 -> (B,T,512) f32
  attn_kernel<<<dim3(BATCH * NHEAD_, TT / 128), 256, 0, stream>>>(qkvT, aT);

  // GN2: (B,T,C) f32 -> (B,T,C) bf16 (reuse xn)
  groupnorm_kernel<<<BATCH * GROUPS_, 256, 0, stream>>>(aT, g2, be2, xn, /*in_is_ct=*/0);

  // proj GEMM + residual: (512x512) x (B,2048,512) -> (B,512,2048) f32 (+x)
  gemm_bf16_kernel<<<dim3(CCH / 128, TT / 64, BATCH), 256, 0, stream>>>(
      wp16, xn, bproj, nullptr, out, x, CCH, TT, CCH);
}
